// Glm4MoeMoE_59133109731630
// MI455X (gfx1250) — compile-verified, matrix operand order checked by reading, not run on hardware
//
#include <hip/hip_runtime.h>
#include <math.h>

typedef __attribute__((ext_vector_type(16))) _Float16 v16h;
typedef __attribute__((ext_vector_type(8)))  _Float16 v8h;
typedef __attribute__((ext_vector_type(8)))  float    v8f;

namespace {
constexpr int TT   = 2048;  // tokens
constexpr int HD   = 1024;  // hidden
constexpr int NE   = 64;    // experts
constexpr int KTOP = 8;     // top-k
constexpr int FF   = 512;   // expert ffn (== shared ffn)
constexpr int CAP  = 512;   // per-expert capacity

constexpr int BM = 128;     // block tile M
constexpr int BN = 64;      // block tile N
constexpr int KB = 32;      // K step (WMMA K)
}

// ---------------------------------------------------------------------------
// Fragment load from an LDS tile laid out [rows][KB] (f16, row-major).
// ISA 16-bit A/B layout: lane&15 selects the row (M or N), lane>>4 selects
// which K-half; elements 0..7 -> K = kb..kb+7, elements 8..15 -> K = kb+16..kb+23.
// ---------------------------------------------------------------------------
__device__ __forceinline__ v16h load_frag(const _Float16* lds, int rowBase, int lane) {
  int r  = rowBase + (lane & 15);
  int kb = (lane >> 4) * 8;
  v8h lo = *(const v8h*)(lds + r * KB + kb);
  v8h hi = *(const v8h*)(lds + r * KB + kb + 16);
  return __builtin_shufflevector(lo, hi, 0,1,2,3,4,5,6,7,8,9,10,11,12,13,14,15);
}

__device__ __forceinline__ v8f wmma_f16(v16h a, v16h b, v8f c) {
  return __builtin_amdgcn_wmma_f32_16x16x32_f16(false, a, false, b, (short)0, c, false, false);
}

// low 32 bits of a generic pointer into __shared__ = LDS byte offset
__device__ __forceinline__ unsigned lds_addr(const void* p) {
  return (unsigned)(uintptr_t)p;
}

// ---------------------------------------------------------------------------
// Router: logits = x @ gate_w^T, softmax over 64 experts, top-8, renormalize.
// One block (64 threads) per token.
// ---------------------------------------------------------------------------
__global__ __launch_bounds__(64) void moe_router_kernel(
    const float* __restrict__ x, const float* __restrict__ gate_w,
    int* __restrict__ topk_idx, float* __restrict__ topk_w)
{
  __shared__ __align__(16) float xs[HD];
  __shared__ float red[NE];
  __shared__ float probs[NE];

  const int t   = blockIdx.x;
  const int tid = threadIdx.x;

  const float4* xrow = (const float4*)(x + (size_t)t * HD);
  float4* xs4 = (float4*)xs;
  for (int i = tid; i < HD / 4; i += 64) xs4[i] = xrow[i];
  __syncthreads();

  const float4* w4 = (const float4*)(gate_w + (size_t)tid * HD);
  float acc = 0.f;
  for (int i = 0; i < HD / 4; ++i) {
    float4 a = xs4[i];
    float4 b = w4[i];
    acc += a.x * b.x + a.y * b.y + a.z * b.z + a.w * b.w;
  }

  red[tid] = acc; __syncthreads();
  for (int s = 32; s > 0; s >>= 1) {
    if (tid < s) red[tid] = fmaxf(red[tid], red[tid + s]);
    __syncthreads();
  }
  const float mx = red[0]; __syncthreads();

  const float ex = __expf(acc - mx);
  red[tid] = ex; __syncthreads();
  for (int s = 32; s > 0; s >>= 1) {
    if (tid < s) red[tid] += red[tid + s];
    __syncthreads();
  }
  probs[tid] = ex * __builtin_amdgcn_rcpf(red[0]);
  __syncthreads();

  if (tid == 0) {
    int   bi[KTOP];
    float bv[KTOP];
    float wsum = 0.f;
    for (int k = 0; k < KTOP; ++k) {
      float best = -1.f; int bidx = 0;
      for (int e = 0; e < NE; ++e) {
        float v = probs[e];
        if (v > best) { best = v; bidx = e; }
      }
      probs[bidx] = -1.f;
      bi[k] = bidx; bv[k] = best; wsum += best;
    }
    const float inv = 1.f / wsum;
    for (int k = 0; k < KTOP; ++k) {
      topk_idx[t * KTOP + k] = bi[k];
      topk_w  [t * KTOP + k] = bv[k] * inv;
    }
  }
}

// ---------------------------------------------------------------------------
// Dispatch: per expert, compact chosen tokens (ascending token order) into
// capacity-C slots. One wave32 per expert; ballot + popc prefix.
// ---------------------------------------------------------------------------
__global__ __launch_bounds__(32) void moe_dispatch_kernel(
    const int* __restrict__ topk_idx, const float* __restrict__ topk_w,
    int* __restrict__ sel_idx, float* __restrict__ sel_w)
{
  const int e    = blockIdx.x;
  const int lane = threadIdx.x;
  int base = 0;
  for (int t0 = 0; t0 < TT; t0 += 32) {
    const int t = t0 + lane;
    float w = 0.f;
#pragma unroll
    for (int k = 0; k < KTOP; ++k)
      if (topk_idx[t * KTOP + k] == e) w = topk_w[t * KTOP + k];
    const unsigned mask = (unsigned)__builtin_amdgcn_ballot_w32(w != 0.f);
    const int off = __popc(mask & ((1u << lane) - 1u));
    if (w != 0.f) {
      const int slot = base + off;
      if (slot < CAP) {
        sel_idx[e * CAP + slot] = t;
        sel_w  [e * CAP + slot] = w;
      }
    }
    base += __popc(mask);
  }
  if (base > CAP) base = CAP;
  for (int s = base + lane; s < CAP; s += 32) {
    sel_idx[e * CAP + s] = 0;
    sel_w  [e * CAP + s] = 0.f;
  }
}

// ---------------------------------------------------------------------------
// Stage A: Hact[e,c,f] = silu(xe·Wg^T) * (xe·Wu^T)  (f16 out, fp32 accum).
// 256 threads (8 waves), BM x BN tile, double-buffered 32-wide K staging:
//   iter: issue next-K global loads -> WMMA on current buffer -> convert/store
//   next buffer -> single barrier.
// sel == nullptr -> identity row mapping (shared expert path).
// ---------------------------------------------------------------------------
__global__ __launch_bounds__(256) void moe_gemm_swiglu_kernel(
    const float* __restrict__ X, const int* __restrict__ sel,
    const float* __restrict__ Wg, const float* __restrict__ Wu,
    _Float16* __restrict__ Hout, size_t wStrideE, size_t hStrideE)
{
  __shared__ __align__(16) _Float16 lA[2][BM * KB];
  __shared__ __align__(16) _Float16 lG[2][BN * KB];
  __shared__ __align__(16) _Float16 lU[2][BN * KB];
  __shared__ int rowIdx[BM];

  const int e    = blockIdx.z;
  const int m0   = blockIdx.y * BM;
  const int n0   = blockIdx.x * BN;
  const int tid  = threadIdx.x;
  const int lane = tid & 31;
  const int wave = tid >> 5;
  const int wm   = (wave & 3) * 32;   // 4 waves along M
  const int wn   = (wave >> 2) * 32;  // 2 waves along N

  const float* wgp = Wg + (size_t)e * wStrideE;
  const float* wup = Wu + (size_t)e * wStrideE;

  if (tid < BM) {
    const int r = m0 + tid;
    rowIdx[tid] = sel ? sel[(size_t)e * CAP + r] : r;
  }
  __syncthreads();

  float4 va[4], vg[2], vu[2];

  auto loadGlobals = [&](int k0) {
#pragma unroll
    for (int i = 0; i < 4; ++i) {
      const int chunk = tid + i * 256;
      const int r  = chunk >> 3;
      const int kc = (chunk & 7) * 4;
      va[i] = *(const float4*)(X + (size_t)rowIdx[r] * HD + k0 + kc);
    }
#pragma unroll
    for (int i = 0; i < 2; ++i) {
      const int chunk = tid + i * 256;
      const int r  = chunk >> 3;
      const int kc = (chunk & 7) * 4;
      const size_t go = (size_t)(n0 + r) * HD + k0 + kc;
      vg[i] = *(const float4*)(wgp + go);
      vu[i] = *(const float4*)(wup + go);
    }
  };

  auto stageLds = [&](int b) {
#pragma unroll
    for (int i = 0; i < 4; ++i) {
      const int chunk = tid + i * 256;
      const int r  = chunk >> 3;
      const int kc = (chunk & 7) * 4;
      _Float16* d = &lA[b][r * KB + kc];
      d[0] = (_Float16)va[i].x; d[1] = (_Float16)va[i].y;
      d[2] = (_Float16)va[i].z; d[3] = (_Float16)va[i].w;
    }
#pragma unroll
    for (int i = 0; i < 2; ++i) {
      const int chunk = tid + i * 256;
      const int r  = chunk >> 3;
      const int kc = (chunk & 7) * 4;
      _Float16* dg = &lG[b][r * KB + kc];
      _Float16* du = &lU[b][r * KB + kc];
      dg[0] = (_Float16)vg[i].x; dg[1] = (_Float16)vg[i].y;
      dg[2] = (_Float16)vg[i].z; dg[3] = (_Float16)vg[i].w;
      du[0] = (_Float16)vu[i].x; du[1] = (_Float16)vu[i].y;
      du[2] = (_Float16)vu[i].z; du[3] = (_Float16)vu[i].w;
    }
  };

  v8f accG[2][2], accU[2][2];
#pragma unroll
  for (int i = 0; i < 2; ++i)
#pragma unroll
    for (int j = 0; j < 2; ++j) { accG[i][j] = (v8f)(0.f); accU[i][j] = (v8f)(0.f); }

  loadGlobals(0);
  stageLds(0);
  __syncthreads();

  for (int k0 = 0; k0 < HD; k0 += KB) {
    const int  cur  = (k0 / KB) & 1;
    const bool more = (k0 + KB) < HD;
    if (more) loadGlobals(k0 + KB);

    const v16h a0 = load_frag(&lA[cur][0], wm,      lane);
    const v16h a1 = load_frag(&lA[cur][0], wm + 16, lane);
    const v16h g0 = load_frag(&lG[cur][0], wn,      lane);
    const v16h g1 = load_frag(&lG[cur][0], wn + 16, lane);
    const v16h u0 = load_frag(&lU[cur][0], wn,      lane);
    const v16h u1 = load_frag(&lU[cur][0], wn + 16, lane);

    accG[0][0] = wmma_f16(a0, g0, accG[0][0]);
    accG[0][1] = wmma_f16(a0, g1, accG[0][1]);
    accG[1][0] = wmma_f16(a1, g0, accG[1][0]);
    accG[1][1] = wmma_f16(a1, g1, accG[1][1]);
    accU[0][0] = wmma_f16(a0, u0, accU[0][0]);
    accU[0][1] = wmma_f16(a0, u1, accU[0][1]);
    accU[1][0] = wmma_f16(a1, u0, accU[1][0]);
    accU[1][1] = wmma_f16(a1, u1, accU[1][1]);

    if (more) stageLds(cur ^ 1);
    __syncthreads();
  }

  // epilogue: silu(g)*u with fast rcp. C/D layout: M = q + 8*(lane>=16), N = lane&15.
  _Float16* hp = Hout + (size_t)e * hStrideE;
  const int mAdd = (lane >> 4) * 8;
  const int nCol = lane & 15;
#pragma unroll
  for (int i2 = 0; i2 < 2; ++i2)
#pragma unroll
    for (int j2 = 0; j2 < 2; ++j2)
#pragma unroll
      for (int q = 0; q < 8; ++q) {
        const float g  = accG[i2][j2][q];
        const float u  = accU[i2][j2][q];
        const float hv = g * __builtin_amdgcn_rcpf(1.f + __expf(-g)) * u;
        const int row = m0 + wm + i2 * 16 + mAdd + q;
        const int col = n0 + wn + j2 * 16 + nCol;
        hp[(size_t)row * FF + col] = (_Float16)hv;
      }
}

// ---------------------------------------------------------------------------
// Stage B: out[tok,h] += scale(row) * (Act · Wd^T)[row,h]  via atomicAdd.
// A is f16 [M,FF]: copied with global_load_async_to_lds_b128 (ASYNCcnt).
// Wd is fp32 [N(=H), FF]: register-prefetch + f16 convert into LDS.
// Double-buffered; one barrier per K step.
// sel == nullptr -> tok = row, scale = 1 (shared expert path).
// ---------------------------------------------------------------------------
__global__ __launch_bounds__(256) void moe_gemm_down_kernel(
    const _Float16* __restrict__ Act, const float* __restrict__ Wd,
    float* __restrict__ out,
    const int* __restrict__ sel, const float* __restrict__ selw,
    size_t aStrideE, size_t wStrideE)
{
  __shared__ __align__(16) _Float16 lA[2][BM * KB];
  __shared__ __align__(16) _Float16 lB[2][BN * KB];

  const int e    = blockIdx.z;
  const int m0   = blockIdx.y * BM;
  const int n0   = blockIdx.x * BN;
  const int tid  = threadIdx.x;
  const int lane = tid & 31;
  const int wave = tid >> 5;
  const int wm   = (wave & 3) * 32;
  const int wn   = (wave >> 2) * 32;

  const _Float16* A  = Act + (size_t)e * aStrideE;
  const float*    wd = Wd  + (size_t)e * wStrideE;
  const unsigned long long abase = (unsigned long long)(uintptr_t)A;

  float4 vb[2];

  // async copy of the f16 A tile straight into LDS (128x32 halves = 512 x 16B)
  auto asyncA = [&](int k0, int b) {
#pragma unroll
    for (int i = 0; i < 2; ++i) {
      const int chunk = tid + i * 256;
      const int r  = chunk >> 2;
      const int kc = (chunk & 3) * 8;
      const unsigned ldsa = lds_addr(&lA[b][r * KB + kc]);
      const unsigned goff =
          (unsigned)(((size_t)(m0 + r) * FF + k0 + kc) * sizeof(_Float16));
      asm volatile("global_load_async_to_lds_b128 %0, %1, %2"
                   :: "v"(ldsa), "v"(goff), "s"(abase)
                   : "memory");
    }
  };

  auto loadB = [&](int k0) {
#pragma unroll
    for (int i = 0; i < 2; ++i) {
      const int chunk = tid + i * 256;
      const int r  = chunk >> 3;
      const int kc = (chunk & 7) * 4;
      vb[i] = *(const float4*)(wd + (size_t)(n0 + r) * FF + k0 + kc);
    }
  };

  auto stageB = [&](int b) {
#pragma unroll
    for (int i = 0; i < 2; ++i) {
      const int chunk = tid + i * 256;
      const int r  = chunk >> 3;
      const int kc = (chunk & 7) * 4;
      _Float16* d = &lB[b][r * KB + kc];
      d[0] = (_Float16)vb[i].x; d[1] = (_Float16)vb[i].y;
      d[2] = (_Float16)vb[i].z; d[3] = (_Float16)vb[i].w;
    }
  };

  v8f acc[2][2];
#pragma unroll
  for (int i = 0; i < 2; ++i)
#pragma unroll
    for (int j = 0; j < 2; ++j) acc[i][j] = (v8f)(0.f);

  asyncA(0, 0);
  loadB(0);
  stageB(0);
  asm volatile("s_wait_asynccnt 0x0" ::: "memory");
  __syncthreads();

  for (int k0 = 0; k0 < FF; k0 += KB) {
    const int  cur  = (k0 / KB) & 1;
    const bool more = (k0 + KB) < FF;
    if (more) {
      asyncA(k0 + KB, cur ^ 1);
      loadB(k0 + KB);
    }

    const v16h a0 = load_frag(&lA[cur][0], wm,      lane);
    const v16h a1 = load_frag(&lA[cur][0], wm + 16, lane);
    const v16h b0 = load_frag(&lB[cur][0], wn,      lane);
    const v16h b1 = load_frag(&lB[cur][0], wn + 16, lane);

    acc[0][0] = wmma_f16(a0, b0, acc[0][0]);
    acc[0][1] = wmma_f16(a0, b1, acc[0][1]);
    acc[1][0] = wmma_f16(a1, b0, acc[1][0]);
    acc[1][1] = wmma_f16(a1, b1, acc[1][1]);

    if (more) stageB(cur ^ 1);
    asm volatile("s_wait_asynccnt 0x0" ::: "memory");
    __syncthreads();
  }

  const int mAdd = (lane >> 4) * 8;
  const int nCol = lane & 15;
#pragma unroll
  for (int i2 = 0; i2 < 2; ++i2)
#pragma unroll
    for (int j2 = 0; j2 < 2; ++j2)
#pragma unroll
      for (int q = 0; q < 8; ++q) {
        const int row = m0 + wm + i2 * 16 + mAdd + q;
        const int col = n0 + wn + j2 * 16 + nCol;
        const float scale = selw ? selw[(size_t)e * CAP + row] : 1.f;
        if (scale != 0.f) {  // skip capacity-padding rows
          const int tok = sel ? sel[(size_t)e * CAP + row] : row;
          atomicAdd(out + (size_t)tok * HD + col, acc[i2][j2][q] * scale);
        }
      }
}

// ---------------------------------------------------------------------------
extern "C" void kernel_launch(void* const* d_in, const int* in_sizes, int n_in,
                              void* d_out, int out_size, void* d_ws, size_t ws_size,
                              hipStream_t stream) {
  (void)in_sizes; (void)n_in; (void)out_size; (void)ws_size;

  const float* x   = (const float*)d_in[0];
  const float* gw  = (const float*)d_in[1];
  const float* wg  = (const float*)d_in[2];
  const float* wu  = (const float*)d_in[3];
  const float* wd  = (const float*)d_in[4];
  const float* sg  = (const float*)d_in[5];
  const float* su  = (const float*)d_in[6];
  const float* sd  = (const float*)d_in[7];
  float* out = (float*)d_out;

  char* ws = (char*)d_ws;
  size_t off = 0;
  auto alloc = [&](size_t bytes) {
    void* p = ws + off;
    off = (off + bytes + 255) & ~(size_t)255;
    return p;
  };
  int*      topk_idx = (int*)     alloc((size_t)TT * KTOP * 4);
  float*    topk_w   = (float*)   alloc((size_t)TT * KTOP * 4);
  int*      sel_idx  = (int*)     alloc((size_t)NE * CAP * 4);
  float*    sel_w    = (float*)   alloc((size_t)NE * CAP * 4);
  _Float16* Hact     = (_Float16*)alloc((size_t)NE * CAP * FF * 2);
  _Float16* Sact     = (_Float16*)alloc((size_t)TT * FF * 2);

  hipMemsetAsync(out, 0, (size_t)TT * HD * sizeof(float), stream);

  moe_router_kernel<<<TT, 64, 0, stream>>>(x, gw, topk_idx, topk_w);
  moe_dispatch_kernel<<<NE, 32, 0, stream>>>(topk_idx, topk_w, sel_idx, sel_w);

  // routed experts: SwiGLU up/gate
  moe_gemm_swiglu_kernel<<<dim3(FF / BN, CAP / BM, NE), 256, 0, stream>>>(
      x, sel_idx, wg, wu, Hact, (size_t)FF * HD, (size_t)CAP * FF);
  // shared expert: SwiGLU up/gate (identity rows)
  moe_gemm_swiglu_kernel<<<dim3(FF / BN, TT / BM, 1), 256, 0, stream>>>(
      x, nullptr, sg, su, Sact, 0, 0);

  // routed experts: down-proj, weighted scatter-add
  moe_gemm_down_kernel<<<dim3(HD / BN, CAP / BM, NE), 256, 0, stream>>>(
      Hact, wd, out, sel_idx, sel_w, (size_t)CAP * FF, (size_t)HD * FF);
  // shared expert: down-proj, direct add
  moe_gemm_down_kernel<<<dim3(HD / BN, TT / BM, 1), 256, 0, stream>>>(
      Sact, sd, out, nullptr, nullptr, 0, 0);
}
